// EGNNDiff_1864015807170
// MI455X (gfx1250) — matrix operand order repaired and checked by reading, not compiled, not run on hardware
//
#include <hip/hip_runtime.h>

// ---------------------------------------------------------------------------
// Types for CDNA5 WMMA (wave32): v_wmma_f32_16x16x32_bf16
// ---------------------------------------------------------------------------
typedef __attribute__((ext_vector_type(16))) __bf16 bf16x16;
typedef __attribute__((ext_vector_type(8)))  float  f32x8;

union BF16Frag { bf16x16 v; uint4 q[2]; };

#define TILE 64          // edges / nodes per workgroup (4 waves x 16 rows)
#define HDIM 128

#if __has_builtin(__builtin_amdgcn_sched_group_barrier)
#define SCHED_GROUP(mask, size) __builtin_amdgcn_sched_group_barrier((mask), (size), 0)
#else
#define SCHED_GROUP(mask, size)
#endif
// masks: 0x008 = MFMA/WMMA, 0x020 = VMEM read, 0x100 = DS read

__device__ __forceinline__ float silu_f(float x) {
  return x * (1.0f / (1.0f + __expf(-x)));
}

// ---------------------------------------------------------------------------
// Core WMMA micro-GEMM:  acc[8] (16 rows x 128 cols) = bias + A(16xK) * Wt^T
//   A  : bf16, row-major in LDS, row stride = lda elements (rows m0..m0+15)
//   Wt : bf16 weights, TRANSPOSED [128][ldw] row-major in global (L2-resident)
//
// Software-pipelined: the GEMM is split into 2*KSTEPS phases of 4 N-tiles.
// While phase p's 4 WMMAs execute, phase p+1's 8 global_load_b128 are already
// in flight (double-buffered fragments). sched_group_barrier pins the
// loads-ahead-of-WMMA interleave so the scheduler cannot re-serialize it.
// ---------------------------------------------------------------------------
template<int KSTEPS>
__device__ __forceinline__ void wmma_gemm(
    const __bf16* A, int lda, int m0,
    const __bf16* __restrict__ Wt, int ldw,
    const float* __restrict__ bias, f32x8* acc, unsigned lane)
{
  const int nloc = lane & 15;
  const int kh   = lane >> 4;
#pragma unroll
  for (int t = 0; t < 8; ++t) {
    float b = bias ? bias[t * 16 + nloc] : 0.0f;
    f32x8 v;
#pragma unroll
    for (int j = 0; j < 8; ++j) v[j] = b;
    acc[t] = v;
  }

  const __bf16* arow = A  + (size_t)(m0 + nloc) * lda + kh * 8;   // A: lane row
  const __bf16* bcol = Wt + (size_t)nloc * ldw + kh * 8;          // B: lane col

  BF16Frag a[2];
  BF16Frag b[2][4];

  // ---- preamble: A(ks=0) + B phase 0 (tiles 0..3 of ks=0)
  a[0].q[0] = *(const uint4*)(arow);
  a[0].q[1] = *(const uint4*)(arow + 16);
#pragma unroll
  for (int tt = 0; tt < 4; ++tt) {
    const __bf16* bp = bcol + (size_t)tt * 16 * ldw;
    b[0][tt].q[0] = *(const uint4*)(bp);
    b[0][tt].q[1] = *(const uint4*)(bp + 16);
  }
  SCHED_GROUP(0x100, 2);   // 2 DS reads   (A frag)
  SCHED_GROUP(0x020, 8);   // 8 VMEM reads (B phase)

#pragma unroll
  for (int ph = 0; ph < 2 * KSTEPS; ++ph) {
    const int ks   = ph >> 1;
    const int half = ph & 1;
    const int nph  = ph + 1;
    // prefetch next phase into the other buffer
    if (nph < 2 * KSTEPS) {
      const int nks = nph >> 1, nhalf = nph & 1;
      if (nhalf == 0) {
        const __bf16* ap = arow + nks * 32;
        a[nks & 1].q[0] = *(const uint4*)(ap);
        a[nks & 1].q[1] = *(const uint4*)(ap + 16);
      }
      const __bf16* bb = bcol + nks * 32 + (size_t)(nhalf * 4) * 16 * ldw;
#pragma unroll
      for (int tt = 0; tt < 4; ++tt) {
        const __bf16* bp = bb + (size_t)tt * 16 * ldw;
        b[nph & 1][tt].q[0] = *(const uint4*)(bp);
        b[nph & 1][tt].q[1] = *(const uint4*)(bp + 16);
      }
      if (nhalf == 0) SCHED_GROUP(0x100, 2);
      SCHED_GROUP(0x020, 8);
    }
    // 4 WMMAs of the current phase (overlap with in-flight next-phase loads)
#pragma unroll
    for (int tt = 0; tt < 4; ++tt) {
      const int t = half * 4 + tt;
      acc[t] = __builtin_amdgcn_wmma_f32_16x16x32_bf16(
          false, a[ks & 1].v, false, b[ph & 1][tt].v, (short)0, acc[t],
          false, false);
    }
    SCHED_GROUP(0x008, 4);  // 4 WMMA
  }
}

// ---------------------------------------------------------------------------
// Preprocessing kernels
// ---------------------------------------------------------------------------
__global__ void k_transpose_bf16(const float* __restrict__ W,
                                 __bf16* __restrict__ Wt, int K, int Nn) {
  int idx = blockIdx.x * blockDim.x + threadIdx.x;
  if (idx >= K * Nn) return;
  int n = idx / K, k = idx - n * K;
  Wt[idx] = (__bf16)W[(size_t)k * Nn + n];
}

__global__ void k_copy_rows_f32(const float* __restrict__ W,
                                float* __restrict__ out,
                                int rowStart, int count, int Nn) {
  int i = blockIdx.x * blockDim.x + threadIdx.x;
  if (i < count) out[i] = W[(size_t)rowStart * Nn + i];
}

__global__ void k_f32_to_bf16(const float* __restrict__ src,
                              __bf16* __restrict__ dst, int n) {
  int i = blockIdx.x * blockDim.x + threadIdx.x;
  if (i < n) dst[i] = (__bf16)src[i];
}

__global__ void k_degree(const long long* __restrict__ cols,
                         float* __restrict__ deg, int E) {
  int i = blockIdx.x * blockDim.x + threadIdx.x;
  if (i < E) atomicAdd(&deg[cols[i]], 1.0f);
}

// ---------------------------------------------------------------------------
// Embedding: h0 = h_in(64) @ W_emb(64x128) + b_emb   (WMMA, K=64)
// ---------------------------------------------------------------------------
__global__ __launch_bounds__(128, 1) void k_embed(
    const __bf16* __restrict__ hin, const __bf16* __restrict__ Wembt,
    const float* __restrict__ bemb,
    float* __restrict__ hf, __bf16* __restrict__ hbf, int Ntot)
{
  __shared__ __bf16 feat[TILE * 64];
  const unsigned tid = threadIdx.x, lane = tid & 31, wave = tid >> 5;
  const int n0 = blockIdx.x * TILE;
  {
    int el = tid >> 1, half = tid & 1;
    int node = n0 + el; if (node >= Ntot) node = Ntot - 1;
    const __bf16* src = hin + (size_t)node * 64 + half * 32;
    __bf16* dst = feat + el * 64 + half * 32;
#pragma unroll
    for (int k = 0; k < 32; k += 8)
      *(uint4*)(dst + k) = *(const uint4*)(src + k);
  }
  __syncthreads();
  const int m0 = wave * 16, nloc = lane & 15, lh = lane >> 4;
  f32x8 acc[8];
  wmma_gemm<2>(feat, 64, m0, Wembt, 64, bemb, acc, lane);
#pragma unroll
  for (int t = 0; t < 8; ++t) {
    int n = t * 16 + nloc;
#pragma unroll
    for (int j = 0; j < 8; ++j) {
      int m = m0 + j + 8 * lh;
      int node = n0 + m;
      if (node < Ntot) {
        float v = acc[t][j];
        hf [(size_t)node * HDIM + n] = v;
        hbf[(size_t)node * HDIM + n] = (__bf16)v;
      }
    }
  }
}

// ---------------------------------------------------------------------------
// Fused per-layer edge kernel: 64 edges per workgroup.
//   m1 = silu([h_r,h_c,dist] @ Wm1 + bm1)       (WMMA K=256 + rank-1 dist)
//   m  = silu(m1 @ Wm2 + bm2)  -> atomic agg[col]
//   c1 = silu(m @ Wc1 + bc1); cm = tanh(c1 . Wc2)
//   atomic xupd[col] += cm * rel/(dist+eps)
// ---------------------------------------------------------------------------
__global__ __launch_bounds__(128, 1) void k_edge(
    const long long* __restrict__ rows, const long long* __restrict__ cols,
    const __bf16* __restrict__ hbf, const float* __restrict__ xcur,
    const __bf16* __restrict__ Wm1t, const float* __restrict__ wm1dist,
    const float* __restrict__ bm1,
    const __bf16* __restrict__ Wm2t, const float* __restrict__ bm2,
    const __bf16* __restrict__ Wc1t, const float* __restrict__ bc1,
    const float* __restrict__ Wc2,
    float* __restrict__ agg, float* __restrict__ xupd, int Etot)
{
  __shared__ __bf16 feat[TILE * 256];
  __shared__ __bf16 mb  [TILE * HDIM];
  __shared__ float  cb  [TILE * HDIM];
  __shared__ float  dist_s[TILE];
  __shared__ float  rel_s[TILE][3];
  __shared__ int    col_s[TILE];
  __shared__ float  wc2s[HDIM];

  const unsigned tid = threadIdx.x, lane = tid & 31, wave = tid >> 5;
  const int e0 = blockIdx.x * TILE;

  wc2s[tid & 127] = Wc2[tid & 127];
  {
    int el = tid >> 1, half = tid & 1;
    int e = e0 + el; if (e >= Etot) e = Etot - 1;
    long long r = rows[e], c = cols[e];
    const __bf16* src = hbf + (size_t)(half ? c : r) * HDIM;
    __bf16* dst = feat + el * 256 + half * HDIM;
#pragma unroll
    for (int k = 0; k < HDIM; k += 8)
      *(uint4*)(dst + k) = *(const uint4*)(src + k);
    if (half == 0) {
      float rx = xcur[r*3+0] - xcur[c*3+0];
      float ry = xcur[r*3+1] - xcur[c*3+1];
      float rz = xcur[r*3+2] - xcur[c*3+2];
      float d = sqrtf(rx*rx + ry*ry + rz*rz);
      dist_s[el] = d;
      rel_s[el][0] = rx; rel_s[el][1] = ry; rel_s[el][2] = rz;
      col_s[el] = (int)c;
    }
  }
  __syncthreads();

  const int m0 = wave * 16, nloc = lane & 15, lh = lane >> 4;
  f32x8 acc[8];

  // ---- GEMM1: K=256 via WMMA + rank-1 dist column, then SiLU
  wmma_gemm<8>(feat, 256, m0, Wm1t, 256, bm1, acc, lane);
#pragma unroll
  for (int t = 0; t < 8; ++t) {
    int n = t * 16 + nloc;
    float wd = wm1dist[n];
#pragma unroll
    for (int j = 0; j < 8; ++j) {
      int m = m0 + j + 8 * lh;
      mb[m * HDIM + n] = (__bf16)silu_f(acc[t][j] + dist_s[m] * wd);
    }
  }

  // ---- GEMM2: m = silu(m1 @ Wm2 + bm2); scatter agg
  wmma_gemm<4>(mb, HDIM, m0, Wm2t, HDIM, bm2, acc, lane);
#pragma unroll
  for (int t = 0; t < 8; ++t) {
    int n = t * 16 + nloc;
#pragma unroll
    for (int j = 0; j < 8; ++j) {
      int m = m0 + j + 8 * lh;
      float v = silu_f(acc[t][j]);
      mb[m * HDIM + n] = (__bf16)v;
      if (e0 + m < Etot)
        atomicAdd(&agg[(size_t)col_s[m] * HDIM + n], v);
    }
  }

  // ---- GEMM3: c1 = silu(m @ Wc1 + bc1)
  wmma_gemm<4>(mb, HDIM, m0, Wc1t, HDIM, bc1, acc, lane);
#pragma unroll
  for (int t = 0; t < 8; ++t) {
    int n = t * 16 + nloc;
#pragma unroll
    for (int j = 0; j < 8; ++j) {
      int m = m0 + j + 8 * lh;
      cb[m * HDIM + n] = silu_f(acc[t][j]);
    }
  }
  __syncthreads();

  if (tid < TILE && (e0 + (int)tid) < Etot) {
    const float* c1 = cb + tid * HDIM;
    float s = 0.f;
    for (int k = 0; k < HDIM; ++k) s += c1[k] * wc2s[k];
    float cm = tanhf(s);
    float inv = 1.f / (dist_s[tid] + 1e-8f);
    size_t ci = (size_t)col_s[tid] * 3;
    atomicAdd(&xupd[ci + 0], cm * rel_s[tid][0] * inv);
    atomicAdd(&xupd[ci + 1], cm * rel_s[tid][1] * inv);
    atomicAdd(&xupd[ci + 2], cm * rel_s[tid][2] * inv);
  }
}

// ---------------------------------------------------------------------------
// Fused per-layer node kernel: x update, node MLP (WMMA), residual, LayerNorm.
// In-place on hf/hbf (each node owned by exactly one tile).
// ---------------------------------------------------------------------------
__global__ __launch_bounds__(128, 1) void k_node(
    float* __restrict__ hf, __bf16* __restrict__ hbf,
    const float* __restrict__ agg, const float* __restrict__ xupd,
    const float* __restrict__ deg, float* __restrict__ xcur,
    const __bf16* __restrict__ Wn1t, const float* __restrict__ bn1,
    const __bf16* __restrict__ Wn2t, const float* __restrict__ bn2,
    const float* __restrict__ lng, const float* __restrict__ lnb, int Ntot)
{
  __shared__ __bf16 feat[TILE * 256];
  __shared__ __bf16 mb  [TILE * HDIM];
  __shared__ float  hb  [TILE * HDIM];
  __shared__ int    nid[TILE];

  const unsigned tid = threadIdx.x, lane = tid & 31, wave = tid >> 5;
  const int n0 = blockIdx.x * TILE;
  {
    int el = tid >> 1, half = tid & 1;
    int node = n0 + el; bool valid = node < Ntot; if (!valid) node = Ntot - 1;
    if (half == 0) {
      nid[el] = node;
      const __bf16* src = hbf + (size_t)node * HDIM;
      __bf16* dst = feat + el * 256;
#pragma unroll
      for (int k = 0; k < HDIM; k += 8)
        *(uint4*)(dst + k) = *(const uint4*)(src + k);
      if (valid) {
        float inv = 1.f / (deg[node] + 1.f);
        xcur[node*3+0] += xupd[node*3+0] * inv;
        xcur[node*3+1] += xupd[node*3+1] * inv;
        xcur[node*3+2] += xupd[node*3+2] * inv;
      }
    } else {
      const float* src = agg + (size_t)node * HDIM;
      __bf16* dst = feat + el * 256 + HDIM;
#pragma unroll
      for (int k = 0; k < HDIM; k += 4) {
        float4 v = *(const float4*)(src + k);
        dst[k+0] = (__bf16)v.x; dst[k+1] = (__bf16)v.y;
        dst[k+2] = (__bf16)v.z; dst[k+3] = (__bf16)v.w;
      }
    }
  }
  __syncthreads();

  const int m0 = wave * 16, nloc = lane & 15, lh = lane >> 4;
  f32x8 acc[8];
  wmma_gemm<8>(feat, 256, m0, Wn1t, 256, bn1, acc, lane);
#pragma unroll
  for (int t = 0; t < 8; ++t) {
    int n = t * 16 + nloc;
#pragma unroll
    for (int j = 0; j < 8; ++j) {
      int m = m0 + j + 8 * lh;
      mb[m * HDIM + n] = (__bf16)silu_f(acc[t][j]);
    }
  }
  wmma_gemm<4>(mb, HDIM, m0, Wn2t, HDIM, bn2, acc, lane);
#pragma unroll
  for (int t = 0; t < 8; ++t) {
    int n = t * 16 + nloc;
#pragma unroll
    for (int j = 0; j < 8; ++j) {
      int m = m0 + j + 8 * lh;
      hb[m * HDIM + n] = hf[(size_t)nid[m] * HDIM + n] + acc[t][j];
    }
  }
  __syncthreads();

  if (tid < TILE && (n0 + (int)tid) < Ntot) {
    int node = n0 + tid;
    const float* r = hb + tid * HDIM;
    float mu = 0.f;
    for (int k = 0; k < HDIM; ++k) mu += r[k];
    mu *= (1.0f / HDIM);
    float var = 0.f;
    for (int k = 0; k < HDIM; ++k) { float d = r[k] - mu; var += d * d; }
    var *= (1.0f / HDIM);
    float rs = rsqrtf(var + 1e-5f);
    for (int k = 0; k < HDIM; ++k) {
      float v = (r[k] - mu) * rs * lng[k] + lnb[k];
      hf [(size_t)node * HDIM + k] = v;
      hbf[(size_t)node * HDIM + k] = (__bf16)v;
    }
  }
}

// ---------------------------------------------------------------------------
// Epsilon-head edge kernel: K=256 WMMA + rank-4 (rel,dist), two more GEMMs,
// per-edge 128x3 projection, scatter eps[col].
// ---------------------------------------------------------------------------
__global__ __launch_bounds__(128, 1) void k_eps_edge(
    const long long* __restrict__ rows, const long long* __restrict__ cols,
    const __bf16* __restrict__ hbf, const float* __restrict__ xcur,
    const __bf16* __restrict__ Wem1t, const float* __restrict__ emExtra,
    const float* __restrict__ bem1,
    const __bf16* __restrict__ Wem2t, const float* __restrict__ bem2,
    const __bf16* __restrict__ Wec1t, const float* __restrict__ bec1,
    const float* __restrict__ Wec2, const float* __restrict__ bec2,
    float* __restrict__ epsout, int Etot)
{
  __shared__ __bf16 feat[TILE * 256];
  __shared__ __bf16 mb  [TILE * HDIM];
  __shared__ float  cb  [TILE * HDIM];
  __shared__ float  dist_s[TILE];
  __shared__ float  rel_s[TILE][3];
  __shared__ int    col_s[TILE];
  __shared__ float  wec2s[HDIM * 3];

  const unsigned tid = threadIdx.x, lane = tid & 31, wave = tid >> 5;
  const int e0 = blockIdx.x * TILE;

  for (int i = tid; i < HDIM * 3; i += 128) wec2s[i] = Wec2[i];
  {
    int el = tid >> 1, half = tid & 1;
    int e = e0 + el; if (e >= Etot) e = Etot - 1;
    long long r = rows[e], c = cols[e];
    const __bf16* src = hbf + (size_t)(half ? c : r) * HDIM;
    __bf16* dst = feat + el * 256 + half * HDIM;
#pragma unroll
    for (int k = 0; k < HDIM; k += 8)
      *(uint4*)(dst + k) = *(const uint4*)(src + k);
    if (half == 0) {
      float rx = xcur[r*3+0] - xcur[c*3+0];
      float ry = xcur[r*3+1] - xcur[c*3+1];
      float rz = xcur[r*3+2] - xcur[c*3+2];
      float d = sqrtf(rx*rx + ry*ry + rz*rz);
      dist_s[el] = d;
      rel_s[el][0] = rx; rel_s[el][1] = ry; rel_s[el][2] = rz;
      col_s[el] = (int)c;
    }
  }
  __syncthreads();

  const int m0 = wave * 16, nloc = lane & 15, lh = lane >> 4;
  f32x8 acc[8];

  wmma_gemm<8>(feat, 256, m0, Wem1t, 256, bem1, acc, lane);
#pragma unroll
  for (int t = 0; t < 8; ++t) {
    int n = t * 16 + nloc;
    float w0 = emExtra[n], w1 = emExtra[HDIM + n];
    float w2 = emExtra[2*HDIM + n], w3 = emExtra[3*HDIM + n];
#pragma unroll
    for (int j = 0; j < 8; ++j) {
      int m = m0 + j + 8 * lh;
      float v = acc[t][j] + rel_s[m][0]*w0 + rel_s[m][1]*w1
                          + rel_s[m][2]*w2 + dist_s[m]*w3;
      mb[m * HDIM + n] = (__bf16)silu_f(v);
    }
  }
  wmma_gemm<4>(mb, HDIM, m0, Wem2t, HDIM, bem2, acc, lane);
#pragma unroll
  for (int t = 0; t < 8; ++t) {
    int n = t * 16 + nloc;
#pragma unroll
    for (int j = 0; j < 8; ++j) {
      int m = m0 + j + 8 * lh;
      mb[m * HDIM + n] = (__bf16)silu_f(acc[t][j]);
    }
  }
  wmma_gemm<4>(mb, HDIM, m0, Wec1t, HDIM, bec1, acc, lane);
#pragma unroll
  for (int t = 0; t < 8; ++t) {
    int n = t * 16 + nloc;
#pragma unroll
    for (int j = 0; j < 8; ++j) {
      int m = m0 + j + 8 * lh;
      cb[m * HDIM + n] = silu_f(acc[t][j]);
    }
  }
  __syncthreads();

  if (tid < TILE && (e0 + (int)tid) < Etot) {
    const float* r = cb + tid * HDIM;
    float s0 = bec2[0], s1 = bec2[1], s2 = bec2[2];
    for (int k = 0; k < HDIM; ++k) {
      float v = r[k];
      s0 += v * wec2s[k*3+0]; s1 += v * wec2s[k*3+1]; s2 += v * wec2s[k*3+2];
    }
    size_t ci = (size_t)col_s[tid] * 3;
    atomicAdd(&epsout[ci+0], s0);
    atomicAdd(&epsout[ci+1], s1);
    atomicAdd(&epsout[ci+2], s2);
  }
}

// ---------------------------------------------------------------------------
// Final node kernel: eps += silu([h,x] @ Wh1 + bh1) @ Wh2 + bh2,
// then write (h, x, eps) tuple into d_out.
// ---------------------------------------------------------------------------
__global__ __launch_bounds__(128, 1) void k_final(
    const float* __restrict__ hf, const __bf16* __restrict__ hbf,
    const float* __restrict__ xcur, const float* __restrict__ epsin,
    const __bf16* __restrict__ Wh1t, const float* __restrict__ hExtra,
    const float* __restrict__ bh1,
    const float* __restrict__ Wh2, const float* __restrict__ bh2,
    float* __restrict__ out, int Ntot)
{
  __shared__ __bf16 feat[TILE * HDIM];
  __shared__ float  cb  [TILE * HDIM];
  __shared__ float  xv  [TILE][3];

  const unsigned tid = threadIdx.x, lane = tid & 31, wave = tid >> 5;
  const int n0 = blockIdx.x * TILE;
  {
    int el = tid >> 1, half = tid & 1;
    int node = n0 + el; if (node >= Ntot) node = Ntot - 1;
    const __bf16* src = hbf + (size_t)node * HDIM + half * 64;
    __bf16* dst = feat + el * HDIM + half * 64;
#pragma unroll
    for (int k = 0; k < 64; k += 8)
      *(uint4*)(dst + k) = *(const uint4*)(src + k);
    if (half == 0) {
      xv[el][0] = xcur[node*3+0];
      xv[el][1] = xcur[node*3+1];
      xv[el][2] = xcur[node*3+2];
    }
  }
  __syncthreads();

  const int m0 = wave * 16, nloc = lane & 15, lh = lane >> 4;
  f32x8 acc[8];
  wmma_gemm<4>(feat, HDIM, m0, Wh1t, HDIM, bh1, acc, lane);
#pragma unroll
  for (int t = 0; t < 8; ++t) {
    int n = t * 16 + nloc;
    float w0 = hExtra[n], w1 = hExtra[HDIM + n], w2 = hExtra[2*HDIM + n];
#pragma unroll
    for (int j = 0; j < 8; ++j) {
      int m = m0 + j + 8 * lh;
      float v = acc[t][j] + xv[m][0]*w0 + xv[m][1]*w1 + xv[m][2]*w2;
      cb[m * HDIM + n] = silu_f(v);
    }
  }
  __syncthreads();

  if (tid < TILE && (n0 + (int)tid) < Ntot) {
    int node = n0 + tid;
    const float* r = cb + tid * HDIM;
    float s0 = bh2[0], s1 = bh2[1], s2 = bh2[2];
    for (int k = 0; k < HDIM; ++k) {
      float v = r[k];
      s0 += v * Wh2[k*3+0]; s1 += v * Wh2[k*3+1]; s2 += v * Wh2[k*3+2];
    }
    // outputs: h [N*128] | x [N*3] | eps [N*3]
    const float* hr = hf + (size_t)node * HDIM;
    float* oh = out + (size_t)node * HDIM;
    for (int k = 0; k < HDIM; ++k) oh[k] = hr[k];
    float* ox = out + (size_t)Ntot * HDIM + (size_t)node * 3;
    ox[0] = xcur[node*3+0]; ox[1] = xcur[node*3+1]; ox[2] = xcur[node*3+2];
    float* oe = out + (size_t)Ntot * (HDIM + 3) + (size_t)node * 3;
    oe[0] = epsin[node*3+0] + s0;
    oe[1] = epsin[node*3+1] + s1;
    oe[2] = epsin[node*3+2] + s2;
  }
}

// ---------------------------------------------------------------------------
// Host orchestration
// ---------------------------------------------------------------------------
extern "C" void kernel_launch(void* const* d_in, const int* in_sizes, int n_in,
                              void* d_out, int out_size, void* d_ws, size_t ws_size,
                              hipStream_t stream) {
  (void)in_sizes; (void)n_in; (void)out_size; (void)ws_size;
  const int N = 50000, E = 640000, L = 3;

  const float*     h_in  = (const float*)d_in[0];
  const float*     x_in  = (const float*)d_in[1];
  const long long* eidx  = (const long long*)d_in[2];
  const long long* rows  = eidx;
  const long long* cols  = eidx + E;
  const float* W_emb = (const float*)d_in[3];  const float* b_emb = (const float*)d_in[4];
  const float* Wm1   = (const float*)d_in[5];  const float* bm1   = (const float*)d_in[6];
  const float* Wm2   = (const float*)d_in[7];  const float* bm2   = (const float*)d_in[8];
  const float* Wc1   = (const float*)d_in[9];  const float* bc1   = (const float*)d_in[10];
  const float* Wc2   = (const float*)d_in[11];
  const float* Wn1   = (const float*)d_in[12]; const float* bn1   = (const float*)d_in[13];
  const float* Wn2   = (const float*)d_in[14]; const float* bn2   = (const float*)d_in[15];
  const float* ln_g  = (const float*)d_in[16]; const float* ln_b  = (const float*)d_in[17];
  const float* Wem1  = (const float*)d_in[18]; const float* bem1  = (const float*)d_in[19];
  const float* Wem2  = (const float*)d_in[20]; const float* bem2  = (const float*)d_in[21];
  const float* Wec1  = (const float*)d_in[22]; const float* bec1  = (const float*)d_in[23];
  const float* Wec2  = (const float*)d_in[24]; const float* bec2  = (const float*)d_in[25];
  const float* Wh1   = (const float*)d_in[26]; const float* bh1   = (const float*)d_in[27];
  const float* Wh2   = (const float*)d_in[28]; const float* bh2   = (const float*)d_in[29];

  // workspace bump allocator
  char* base = (char*)d_ws; size_t off = 0;
  auto alloc = [&](size_t bytes) -> void* {
    off = (off + 255) & ~(size_t)255;
    void* p = base + off; off += bytes; return p;
  };

  __bf16* Wm1t   = (__bf16*)alloc((size_t)L * 128 * 256 * 2);
  float*  wm1d   = (float*) alloc((size_t)L * 128 * 4);
  __bf16* Wm2t   = (__bf16*)alloc((size_t)L * 128 * 128 * 2);
  __bf16* Wc1t   = (__bf16*)alloc((size_t)L * 128 * 128 * 2);
  __bf16* Wn1t   = (__bf16*)alloc((size_t)L * 128 * 256 * 2);
  __bf16* Wn2t   = (__bf16*)alloc((size_t)L * 128 * 128 * 2);
  __bf16* Wembt  = (__bf16*)alloc((size_t)128 * 64 * 2);
  __bf16* Wem1t  = (__bf16*)alloc((size_t)128 * 256 * 2);
  float*  emEx   = (float*) alloc((size_t)4 * 128 * 4);
  __bf16* Wem2t  = (__bf16*)alloc((size_t)128 * 128 * 2);
  __bf16* Wec1t  = (__bf16*)alloc((size_t)128 * 128 * 2);
  __bf16* Wh1t   = (__bf16*)alloc((size_t)128 * 128 * 2);
  float*  hEx    = (float*) alloc((size_t)3 * 128 * 4);
  __bf16* hinbf  = (__bf16*)alloc((size_t)N * 64 * 2);
  float*  hf     = (float*) alloc((size_t)N * 128 * 4);
  __bf16* hbf    = (__bf16*)alloc((size_t)N * 128 * 2);
  float*  xcur   = (float*) alloc((size_t)N * 3 * 4);
  float*  agg    = (float*) alloc((size_t)N * 128 * 4);
  float*  xupd   = (float*) alloc((size_t)N * 3 * 4);
  float*  deg    = (float*) alloc((size_t)N * 4);
  float*  epsb   = (float*) alloc((size_t)N * 3 * 4);

  auto T = [&](const float* W, __bf16* Wt, int K, int Nn) {
    int total = K * Nn;
    k_transpose_bf16<<<(total + 255) / 256, 256, 0, stream>>>(W, Wt, K, Nn);
  };

  for (int i = 0; i < L; ++i) {
    T(Wm1 + (size_t)i * 257 * 128, Wm1t + (size_t)i * 128 * 256, 256, 128);
    k_copy_rows_f32<<<1, 128, 0, stream>>>(Wm1 + (size_t)i * 257 * 128,
                                           wm1d + (size_t)i * 128, 256, 128, 128);
    T(Wm2 + (size_t)i * 128 * 128, Wm2t + (size_t)i * 128 * 128, 128, 128);
    T(Wc1 + (size_t)i * 128 * 128, Wc1t + (size_t)i * 128 * 128, 128, 128);
    T(Wn1 + (size_t)i * 256 * 128, Wn1t + (size_t)i * 128 * 256, 256, 128);
    T(Wn2 + (size_t)i * 128 * 128, Wn2t + (size_t)i * 128 * 128, 128, 128);
  }
  T(W_emb, Wembt, 64, 128);
  T(Wem1,  Wem1t, 256, 128);
  k_copy_rows_f32<<<2, 256, 0, stream>>>(Wem1, emEx, 256, 512, 128);
  T(Wem2, Wem2t, 128, 128);
  T(Wec1, Wec1t, 128, 128);
  T(Wh1,  Wh1t,  128, 128);
  k_copy_rows_f32<<<2, 256, 0, stream>>>(Wh1, hEx, 128, 384, 128);

  k_f32_to_bf16<<<(N * 64 + 255) / 256, 256, 0, stream>>>(h_in, hinbf, N * 64);
  hipMemcpyAsync(xcur, x_in, (size_t)N * 3 * 4, hipMemcpyDeviceToDevice, stream);
  hipMemsetAsync(deg, 0, (size_t)N * 4, stream);
  k_degree<<<(E + 255) / 256, 256, 0, stream>>>(cols, deg, E);

  const int nodeBlocks = (N + TILE - 1) / TILE;
  const int edgeBlocks = (E + TILE - 1) / TILE;

  k_embed<<<nodeBlocks, 128, 0, stream>>>(hinbf, Wembt, b_emb, hf, hbf, N);

  for (int i = 0; i < L; ++i) {
    hipMemsetAsync(agg,  0, (size_t)N * 128 * 4, stream);
    hipMemsetAsync(xupd, 0, (size_t)N * 3 * 4, stream);
    k_edge<<<edgeBlocks, 128, 0, stream>>>(
        rows, cols, hbf, xcur,
        Wm1t + (size_t)i * 128 * 256, wm1d + (size_t)i * 128, bm1 + (size_t)i * 128,
        Wm2t + (size_t)i * 128 * 128, bm2 + (size_t)i * 128,
        Wc1t + (size_t)i * 128 * 128, bc1 + (size_t)i * 128,
        Wc2 + (size_t)i * 128, agg, xupd, E);
    k_node<<<nodeBlocks, 128, 0, stream>>>(
        hf, hbf, agg, xupd, deg, xcur,
        Wn1t + (size_t)i * 128 * 256, bn1 + (size_t)i * 128,
        Wn2t + (size_t)i * 128 * 128, bn2 + (size_t)i * 128,
        ln_g + (size_t)i * 128, ln_b + (size_t)i * 128, N);
  }

  hipMemsetAsync(epsb, 0, (size_t)N * 3 * 4, stream);
  k_eps_edge<<<edgeBlocks, 128, 0, stream>>>(
      rows, cols, hbf, xcur,
      Wem1t, emEx, bem1, Wem2t, bem2, Wec1t, bec1, Wec2, bec2, epsb, E);
  k_final<<<nodeBlocks, 128, 0, stream>>>(
      hf, hbf, xcur, epsb, Wh1t, hEx, bh1, Wh2, bh2, (float*)d_out, N);
}
